// SelfAttention_83588653514770
// MI455X (gfx1250) — matrix operand order repaired and checked by reading, not compiled, not run on hardware
//
#include <hip/hip_runtime.h>
#include <stdint.h>
#include <stddef.h>

typedef __bf16 bf16;
typedef __attribute__((ext_vector_type(16))) bf16  v16bf;
typedef __attribute__((ext_vector_type(8)))  bf16  bf16x8;
typedef __attribute__((ext_vector_type(4)))  bf16  bf16x4;
typedef __attribute__((ext_vector_type(8)))  float v8f;
typedef __attribute__((ext_vector_type(4)))  unsigned int u32x4;
typedef __attribute__((ext_vector_type(4)))  int   i32x4;
typedef __attribute__((ext_vector_type(8)))  int   i32x8;

#define S_LEN   2048
#define D_DIM   768
#define KT      32      // keys per iteration
#define NC      24      // 768/32 k-chunks for scores
#define NDT     24      // 16-wide D tiles per wave (half of 768/16)
#define XPITCH  780     // dwords per x-tile row: 768 + 3 pads * 4 dwords (TDM pad: 4dw / 256dw)
#define VT_PITCH 40     // bf16 per row of transposed tile (768 rows x 32 cols)
#define QPITCH  776     // bf16 per Q row
#define PPITCH  40      // bf16 per P row

struct Smem {
  float xraw[32 * XPITCH];        //  99,840 B : raw f32 K-tile (TDM-padded layout)
  bf16  vt[D_DIM * VT_PITCH];     //  61,440 B : bf16 transposed tile (d-major)
  bf16  qs[4][16 * QPITCH];       //  99,328 B : bf16 Q, 4 row-groups x 16 rows
  bf16  p[8][16 * PPITCH];        //  10,240 B : per-wave P scratch
};                                // total ~270,848 B (< 320 KB WGP LDS)

// padded x-tile index: row t, element d (pads every 256 dwords, 4 dwords each)
__device__ __forceinline__ int xoff(int d) { return ((d >> 8) * 260) + (d & 255); }

// assemble a 16-element bf16 fragment from two contiguous 16-byte runs in LDS
__device__ __forceinline__ v16bf frag2(const bf16* lo, const bf16* hi) {
  bf16x8 a = *(const bf16x8*)lo;
  bf16x8 b = *(const bf16x8*)hi;
  v16bf r;
#pragma unroll
  for (int i = 0; i < 8; ++i) { r[i] = a[i]; r[i + 8] = b[i]; }
  return r;
}

// 16 contiguous f32 in LDS -> bf16 fragment
__device__ __forceinline__ v16bf cvtfrag(const float* s) {
  float4 f0 = *(const float4*)(s);
  float4 f1 = *(const float4*)(s + 4);
  float4 f2 = *(const float4*)(s + 8);
  float4 f3 = *(const float4*)(s + 12);
  v16bf r;
  r[0]  = (bf16)f0.x; r[1]  = (bf16)f0.y; r[2]  = (bf16)f0.z; r[3]  = (bf16)f0.w;
  r[4]  = (bf16)f1.x; r[5]  = (bf16)f1.y; r[6]  = (bf16)f1.z; r[7]  = (bf16)f1.w;
  r[8]  = (bf16)f2.x; r[9]  = (bf16)f2.y; r[10] = (bf16)f2.z; r[11] = (bf16)f2.w;
  r[12] = (bf16)f3.x; r[13] = (bf16)f3.y; r[14] = (bf16)f3.z; r[15] = (bf16)f3.w;
  return r;
}

__global__ __launch_bounds__(256, 1)
void fattn_xxt(const float* __restrict__ xg, float* __restrict__ outg) {
  __shared__ Smem sm;
  const int tid  = threadIdx.x;
  const int w    = tid >> 5;
  const int lane = tid & 31;
  const int n    = lane & 15;
  const int hi   = lane >> 4;
  const int rg   = w >> 1;        // row-group 0..3 (16 rows each)
  const int dh   = w & 1;         // D half 0..1 (384 columns each)
  const int b    = blockIdx.x >> 5;
  const int rowBase = (blockIdx.x & 31) * 64;
  const size_t xbase = (size_t)b * S_LEN * D_DIM;

  // ---- Q prologue: 64 rows x 768 f32 -> bf16 in LDS ----
#pragma unroll 1
  for (int i = 0; i < 48; ++i) {
    int g   = tid + i * 256;
    int row = g / 192;            // 0..63
    int c4  = g % 192;            // float4 index within row
    const float4 f = *(const float4*)(xg + xbase + (size_t)(rowBase + row) * D_DIM + c4 * 4);
    bf16x4 v;
    v[0] = (bf16)f.x; v[1] = (bf16)f.y; v[2] = (bf16)f.z; v[3] = (bf16)f.w;
    *(bf16x4*)&sm.qs[row >> 4][(row & 15) * QPITCH + c4 * 4] = v;
  }

  v8f acc[NDT];
  float mstate[8], lstate[8];
#pragma unroll
  for (int dt = 0; dt < NDT; ++dt)
#pragma unroll
    for (int r = 0; r < 8; ++r) acc[dt][r] = 0.0f;
#pragma unroll
  for (int r = 0; r < 8; ++r) { mstate[r] = -1.0e30f; lstate[r] = 0.0f; }

#pragma unroll 1
  for (int it = 0; it < S_LEN / KT; ++it) {
    const int t0 = it * KT;
    __syncthreads();   // previous tile fully consumed

    // ---- TDM: DMA 32 x 768 f32 tile into LDS (wave 0 only) ----
    if (w == 0) {
      uint64_t ga = (uint64_t)(uintptr_t)(xg + xbase + (size_t)t0 * D_DIM);
      u32x4 g0; i32x8 g1; i32x4 gz4 = {0, 0, 0, 0};
      i32x8 gz8 = {0, 0, 0, 0, 0, 0, 0, 0};
      g0[0] = 1u;                                   // count=1, user descriptor
      g0[1] = 0u;                                   // lds_addr: sm.xraw at LDS offset 0
      g0[2] = (unsigned)ga;                         // global addr lo
      g0[3] = (unsigned)((ga >> 32) & 0x1ffffffu) | (2u << 30);  // addr hi | type=2
      // data_size=4B(2), pad_enable, pad_interval=256dw(7), pad_amount=4dw(3)
      g1[0] = (int)((2u << 16) | (1u << 20) | (7u << 22) | (3u << 25));
      g1[1] = (int)(768u << 16);                    // tensor_dim0 lo16 in [31:16]
      g1[2] = (int)(2048u << 16);                   // tensor_dim0 hi=0 | tensor_dim1 lo16
      g1[3] = (int)(768u << 16);                    // tensor_dim1 hi=0 | tile_dim0=768
      g1[4] = 32;                                   // tile_dim1=32 rows, tile_dim2=0
      g1[5] = 768;                                  // tensor_dim0_stride lo32
      g1[6] = 0;                                    // stride hi | dim1_stride lo
      g1[7] = 0;
      __builtin_amdgcn_tensor_load_to_lds(g0, g1, gz4, gz4, gz8, 0);
      __builtin_amdgcn_s_wait_tensorcnt((short)0);
    }
    __syncthreads();   // xraw visible to all waves

    // ---- build bf16 transposed copy vt[d][t] ----
#pragma unroll 1
    for (int i = 0; i < 12; ++i) {
      int g = tid + i * 256;      // 32 rows * 96 groups of 8
      int t = g / 96, q = g % 96;
      const float* src = &sm.xraw[t * XPITCH + xoff(q * 8)];
      float4 f0 = *(const float4*)src;
      float4 f1 = *(const float4*)(src + 4);
      bf16* dst = &sm.vt[(q * 8) * VT_PITCH + t];
      dst[0 * VT_PITCH] = (bf16)f0.x; dst[1 * VT_PITCH] = (bf16)f0.y;
      dst[2 * VT_PITCH] = (bf16)f0.z; dst[3 * VT_PITCH] = (bf16)f0.w;
      dst[4 * VT_PITCH] = (bf16)f1.x; dst[5 * VT_PITCH] = (bf16)f1.y;
      dst[6 * VT_PITCH] = (bf16)f1.z; dst[7 * VT_PITCH] = (bf16)f1.w;
    }
    __syncthreads();

    // ---- scores: S[16][32] = Q(16x768) x Ktile^T ----
    v8f s0, s1;
#pragma unroll
    for (int r = 0; r < 8; ++r) { s0[r] = 0.0f; s1[r] = 0.0f; }
    const bf16*  qrow = &sm.qs[rg][n * QPITCH];       // A row m = lane&15
    const float* x0   = &sm.xraw[n * XPITCH];         // B tile0: key row n
    const float* x1   = &sm.xraw[(16 + n) * XPITCH];  // B tile1: key row 16+n
#pragma unroll
    for (int c = 0; c < NC; ++c) {
      const bf16* qb = qrow + c * 32 + hi * 16;
      v16bf a  = frag2(qb, qb + 8);
      int   db = c * 32 + hi * 16;
      v16bf b0 = cvtfrag(x0 + xoff(db));
      v16bf b1 = cvtfrag(x1 + xoff(db));
      s0 = __builtin_amdgcn_wmma_f32_16x16x32_bf16(false, a, false, b0, (short)0, s0, false, false);
      s1 = __builtin_amdgcn_wmma_f32_16x16x32_bf16(false, a, false, b1, (short)0, s1, false, false);
    }

    // ---- online softmax (rows M = r + 8*hi, N across lanes in 16-lane halves) ----
    float rmax[8], p0[8], p1[8], scale[8], rs[8];
#pragma unroll
    for (int r = 0; r < 8; ++r) rmax[r] = fmaxf(s0[r], s1[r]);
#pragma unroll
    for (int msk = 8; msk >= 1; msk >>= 1)
#pragma unroll
      for (int r = 0; r < 8; ++r) rmax[r] = fmaxf(rmax[r], __shfl_xor(rmax[r], msk, 32));
#pragma unroll
    for (int r = 0; r < 8; ++r) {
      float mn = fmaxf(mstate[r], rmax[r]);
      scale[r] = __expf(mstate[r] - mn);
      p0[r] = __expf(s0[r] - mn);
      p1[r] = __expf(s1[r] - mn);
      mstate[r] = mn;
      rs[r] = p0[r] + p1[r];
    }
#pragma unroll
    for (int msk = 8; msk >= 1; msk >>= 1)
#pragma unroll
      for (int r = 0; r < 8; ++r) rs[r] += __shfl_xor(rs[r], msk, 32);
#pragma unroll
    for (int r = 0; r < 8; ++r) lstate[r] = lstate[r] * scale[r] + rs[r];
#pragma unroll
    for (int dt = 0; dt < NDT; ++dt)
#pragma unroll
      for (int r = 0; r < 8; ++r) acc[dt][r] *= scale[r];

    // ---- P: C-layout -> LDS -> A-layout fragment (same-wave LDS is in-order) ----
    bf16* pw = sm.p[w];
#pragma unroll
    for (int r = 0; r < 8; ++r) {
      pw[(r + 8 * hi) * PPITCH + n]      = (bf16)p0[r];
      pw[(r + 8 * hi) * PPITCH + 16 + n] = (bf16)p1[r];
    }
    const bf16* pb = &pw[n * PPITCH + hi * 16];
    v16bf pa = frag2(pb, pb + 8);

    // ---- PV: acc[16][dh*384 .. +384) += P(16x32) x Vtile ----
#pragma unroll
    for (int dt = 0; dt < NDT; ++dt) {
      const bf16* vrow = &sm.vt[(dh * 384 + dt * 16 + n) * VT_PITCH + hi * 16];
      v16bf bv = frag2(vrow, vrow + 8);
      acc[dt] = __builtin_amdgcn_wmma_f32_16x16x32_bf16(false, pa, false, bv, (short)0, acc[dt], false, false);
    }
  }

  // ---- epilogue: o = acc / l, out = o * x ----
  float rl[8];
#pragma unroll
  for (int r = 0; r < 8; ++r) rl[r] = 1.0f / lstate[r];
#pragma unroll
  for (int dt = 0; dt < NDT; ++dt) {
#pragma unroll
    for (int r = 0; r < 8; ++r) {
      size_t idx = xbase + (size_t)(rowBase + rg * 16 + r + 8 * hi) * D_DIM
                 + dh * 384 + dt * 16 + n;
      outg[idx] = acc[dt][r] * rl[r] * xg[idx];
    }
  }
}

extern "C" void kernel_launch(void* const* d_in, const int* in_sizes, int n_in,
                              void* d_out, int out_size, void* d_ws, size_t ws_size,
                              hipStream_t stream) {
  (void)in_sizes; (void)n_in; (void)out_size; (void)d_ws; (void)ws_size;
  const float* x = (const float*)d_in[0];
  float* out = (float*)d_out;
  dim3 grid(8 * (S_LEN / 64));   // 256 blocks: 8 batches x 32 row-tiles
  dim3 block(256);               // 8 waves
  hipLaunchKernelGGL(fattn_xxt, grid, block, 0, stream, x, out);
}
